// VanillaPlanHead2_66640712565435
// MI455X (gfx1250) — matrix operand order repaired and checked by reading, not compiled, not run on hardware
//
#include <hip/hip_runtime.h>
#include <hip/hip_bf16.h>
#include <stdint.h>

// ---------------------------------------------------------------------------
// VanillaPlanHead2 fused MLP (21->512->512->21) + segment-weighted L1 loss
// CDNA5 / gfx1250: wave32, v_wmma_f32_16x16x32_bf16 for all three GEMMs.
// Tiling: 64 batch rows / block, 8 waves; each wave = 4 m-tiles x 4 n-tiles
// so every B fragment is reused by 4 WMMAs and fetched once per block.
// ---------------------------------------------------------------------------

typedef __attribute__((ext_vector_type(16))) __bf16 v16bf;
typedef __attribute__((ext_vector_type(8)))  float  v8f;
typedef __attribute__((ext_vector_type(4)))  unsigned int u32x4;

union Frag {
  v16bf bf;
  u32x4 q[2];
};

#define BATCH   131072
#define IN_DIM  21
#define HID     512
#define OUT_DIM 21

__device__ __forceinline__ uint16_t f2bf(float x) {
  union { __bf16 h; uint16_t u; } c;
  c.h = (__bf16)x;                       // native f32->bf16 cvt
  return c.u;
}

// ---------------------------- prep kernels ---------------------------------

__global__ void zero_acc_kernel(float* __restrict__ acc) {
  if (threadIdx.x < 2) acc[threadIdx.x] = 0.f;
}

// velocities [B,21] f32 -> Vp [B,32] bf16 (K padded with zeros)
__global__ void prep_vel_kernel(const float* __restrict__ vel, uint16_t* __restrict__ Vp) {
  int idx = blockIdx.x * blockDim.x + threadIdx.x;       // B*32 elements
  int row = idx >> 5, k = idx & 31;
  Vp[idx] = (k < IN_DIM) ? f2bf(vel[row * IN_DIM + k]) : (uint16_t)0;
}

// W1 [512,21] -> B-fragment order, 32 n-tiles x 1 k-tile, K padded 21->32.
// Fragment halves: lane<16 holds K 0..7,16..23, lane>=16 holds K 8..15,24..31, col = lane%16.
__global__ void prep_w1_kernel(const float* __restrict__ W1, uint16_t* __restrict__ W1p) {
  int e = blockIdx.x * blockDim.x + threadIdx.x;         // 16384 elements
  int nt = e >> 9; int r = e & 511; int l = r >> 4; int j = r & 15;
  int n = nt * 16 + (l & 15);
  int k = (l >> 4) * 8 + (j & 7) + (j >> 3) * 16;
  W1p[e] = (k < IN_DIM) ? f2bf(W1[n * IN_DIM + k]) : (uint16_t)0;
}

// W2 [512,512] -> 16 k-tiles x 32 n-tiles of B fragments (512 halves each).
__global__ void prep_w2_kernel(const float* __restrict__ W2, uint16_t* __restrict__ W2p) {
  int e = blockIdx.x * blockDim.x + threadIdx.x;         // 262144 elements
  int f = e >> 9; int kt = f >> 5; int nt = f & 31;
  int l = (e >> 4) & 31; int j = e & 15;
  int n = nt * 16 + (l & 15);
  int k = kt * 32 + (l >> 4) * 8 + (j & 7) + (j >> 3) * 16;
  W2p[e] = f2bf(W2[n * HID + k]);
}

// W3 [21,512] -> 16 k-tiles x 2 n-tiles (N padded 21->32 with zeros).
__global__ void prep_w3_kernel(const float* __restrict__ W3, uint16_t* __restrict__ W3p) {
  int e = blockIdx.x * blockDim.x + threadIdx.x;         // 16384 elements
  int f = e >> 9; int kt = f >> 1; int nt = f & 1;
  int l = (e >> 4) & 31; int j = e & 15;
  int n = nt * 16 + (l & 15);
  int k = kt * 32 + (l >> 4) * 8 + (j & 7) + (j >> 3) * 16;
  W3p[e] = (n < OUT_DIM) ? f2bf(W3[n * HID + k]) : (uint16_t)0;
}

// ---------------------------- main fused kernel ----------------------------
// Dynamic LDS: hA/hB = 64x512 bf16 each (128 KB total; CDNA5 WGP has 320 KB).

__launch_bounds__(256, 1)
__global__ void plan_head_kernel(const uint16_t* __restrict__ Vp,
                                 const uint16_t* __restrict__ W1p,
                                 const uint16_t* __restrict__ W2p,
                                 const uint16_t* __restrict__ W3p,
                                 const float* __restrict__ b1,
                                 const float* __restrict__ b2,
                                 const float* __restrict__ b3,
                                 const float* __restrict__ gt,
                                 float* __restrict__ acc) {
  extern __shared__ uint16_t smem[];
  uint16_t* hA = smem;                   // 64*512 bf16
  uint16_t* hB = smem + 64 * HID;        // 64*512 bf16

  const int tid  = threadIdx.x;
  const int wid  = tid >> 5;             // 0..7
  const int lane = tid & 31;
  const int l16  = lane & 15;
  const int hi   = lane >> 4;            // 0 or 1
  const int wgRow = blockIdx.x << 6;     // 64 rows per block

  const v8f vzero = {};

  // ---------------- layer 1: Vp[64-row tile] x W1p -> hA ----------------
  {
    Frag a[4];
#pragma unroll
    for (int mt = 0; mt < 4; ++mt) {
      const int row = wgRow + mt * 16 + l16;
      const u32x4* vp = (const u32x4*)(Vp + (size_t)row * 32);
      a[mt].q[0] = vp[hi];               // halves hi*8..
      a[mt].q[1] = vp[hi + 2];           // halves 16+hi*8..
    }
    v8f c[4][4];
#pragma unroll
    for (int t = 0; t < 4; ++t) {
      const int nt = wid * 4 + t;
      Frag b;
      const u32x4* bp = (const u32x4*)(W1p + (size_t)nt * 512);
      b.q[0] = bp[lane * 2];
      b.q[1] = bp[lane * 2 + 1];
#pragma unroll
      for (int mt = 0; mt < 4; ++mt)
        c[mt][t] = __builtin_amdgcn_wmma_f32_16x16x32_bf16(
            false, a[mt].bf, false, b.bf, (short)0, vzero, false, false);
    }
#pragma unroll
    for (int t = 0; t < 4; ++t) {
      const int n = (wid * 4 + t) * 16 + l16;
      const float bias = b1[n];
#pragma unroll
      for (int mt = 0; mt < 4; ++mt)
#pragma unroll
        for (int r = 0; r < 8; ++r) {
          const int m = mt * 16 + hi * 8 + r;   // C layout: lanes 16-31 -> rows +8
          float v = c[mt][t][r] + bias;
          v = v > 0.f ? v : 0.f;
          hA[m * HID + n] = f2bf(v);
        }
    }
  }
  __syncthreads();

  // ---------------- layer 2: hA x W2p -> hB (dominant GEMM) ----------------
  {
    v8f c[4][4];
#pragma unroll
    for (int mt = 0; mt < 4; ++mt)
#pragma unroll
      for (int t = 0; t < 4; ++t) c[mt][t] = vzero;

    for (int kt = 0; kt < 16; ++kt) {
      Frag a[4];
#pragma unroll
      for (int mt = 0; mt < 4; ++mt) {
        const uint16_t* ap = hA + (mt * 16 + l16) * HID + kt * 32 + hi * 8;
        a[mt].q[0] = *(const u32x4*)ap;
        a[mt].q[1] = *(const u32x4*)(ap + 16);
      }
      if (kt < 15)  // pull next k-slab of this wave's column group toward L2/L0
        __builtin_prefetch(W2p + ((size_t)(kt + 1) * 32 + wid * 4) * 512 + lane * 16, 0, 1);
#pragma unroll
      for (int t = 0; t < 4; ++t) {
        const int nt = wid * 4 + t;
        Frag b;
        const u32x4* bp = (const u32x4*)(W2p + (size_t)(kt * 32 + nt) * 512);
        b.q[0] = bp[lane * 2];
        b.q[1] = bp[lane * 2 + 1];
#pragma unroll
        for (int mt = 0; mt < 4; ++mt)
          c[mt][t] = __builtin_amdgcn_wmma_f32_16x16x32_bf16(
              false, a[mt].bf, false, b.bf, (short)0, c[mt][t], false, false);
      }
    }
#pragma unroll
    for (int t = 0; t < 4; ++t) {
      const int n = (wid * 4 + t) * 16 + l16;
      const float bias = b2[n];
#pragma unroll
      for (int mt = 0; mt < 4; ++mt)
#pragma unroll
        for (int r = 0; r < 8; ++r) {
          const int m = mt * 16 + hi * 8 + r;
          float v = c[mt][t][r] + bias;
          v = v > 0.f ? v : 0.f;
          hB[m * HID + n] = f2bf(v);
        }
    }
  }
  __syncthreads();

  // ---------------- layer 3 + fused weighted-L1 loss ----------------
  // 4 m-tiles x 2 padded n-tiles = 8 jobs -> one per wave.
  float num = 0.f, den = 0.f;
  {
    const int mt3 = wid & 3;
    const int nt3 = wid >> 2;              // N padded to 32 -> 2 n-tiles
    const int n = nt3 * 16 + l16;
    v8f c = vzero;
    for (int kt = 0; kt < 16; ++kt) {
      Frag a;
      const uint16_t* ap = hB + (mt3 * 16 + l16) * HID + kt * 32 + hi * 8;
      a.q[0] = *(const u32x4*)ap;
      a.q[1] = *(const u32x4*)(ap + 16);
      Frag b;
      const u32x4* bp = (const u32x4*)(W3p + (size_t)(kt * 2 + nt3) * 512);
      b.q[0] = bp[lane * 2];
      b.q[1] = bp[lane * 2 + 1];
      c = __builtin_amdgcn_wmma_f32_16x16x32_bf16(
          false, a.bf, false, b.bf, (short)0, c, false, false);
    }
    if (n < OUT_DIM) {
      const float bias = b3[n];
#pragma unroll
      for (int r = 0; r < 8; ++r) {
        const int m = wgRow + mt3 * 16 + hi * 8 + r;
        const float g = gt[(size_t)m * OUT_DIM + n];
        if (g > -5000.f) {
          const float p = c[r] + bias;
          // floor(p/0.5) == floor(g/0.5)  <=>  floor(2p) == floor(2g)
          const float w = (floorf(p * 2.f) == floorf(g * 2.f)) ? 0.5f : 1.f;
          num += fabsf((p - g) * w);
          den += 1.f;
        }
      }
    }
#pragma unroll
    for (int off = 16; off > 0; off >>= 1) {
      num += __shfl_down(num, off, 32);
      den += __shfl_down(den, off, 32);
    }
  }
  // block reduction reusing LDS (dead after layer-3 reads complete per wave)
  float* red = (float*)smem;
  if (lane == 0) { red[wid] = num; red[8 + wid] = den; }
  __syncthreads();
  if (tid == 0) {
    float N = 0.f, D = 0.f;
#pragma unroll
    for (int i = 0; i < 8; ++i) { N += red[i]; D += red[8 + i]; }
    atomicAdd(&acc[0], N);
    atomicAdd(&acc[1], D);
  }
}

__global__ void finalize_kernel(const float* __restrict__ acc, float* __restrict__ out) {
  out[0] = acc[0] / acc[1];
}

// ------------------------------- launcher ----------------------------------

extern "C" void kernel_launch(void* const* d_in, const int* in_sizes, int n_in,
                              void* d_out, int out_size, void* d_ws, size_t ws_size,
                              hipStream_t stream) {
  const float* vel = (const float*)d_in[0];
  const float* gt  = (const float*)d_in[1];
  const float* W1  = (const float*)d_in[2];
  const float* b1  = (const float*)d_in[3];
  const float* W2  = (const float*)d_in[4];
  const float* b2  = (const float*)d_in[5];
  const float* W3  = (const float*)d_in[6];
  const float* b3  = (const float*)d_in[7];
  float* out = (float*)d_out;

  // workspace layout (bytes):
  //   [0,256)   float acc[2] (num, den)
  //   Vp   : BATCH*32 bf16 (8 MB)    velocities, K padded to 32
  //   W1p  : 32*512  bf16 (32 KB)    B-fragment order
  //   W2p  : 512*512 bf16 (512 KB)   B-fragment order
  //   W3p  : 32*512  bf16 (32 KB)    B-fragment order, N padded to 32
  float* acc    = (float*)d_ws;
  uint16_t* Vp  = (uint16_t*)((char*)d_ws + 256);
  uint16_t* W1p = Vp + (size_t)BATCH * 32;
  uint16_t* W2p = W1p + 32 * 512;
  uint16_t* W3p = W2p + 512 * 512;

  const size_t ldsBytes = (size_t)2 * 64 * HID * sizeof(uint16_t);  // 128 KB

  zero_acc_kernel<<<1, 64, 0, stream>>>(acc);
  prep_vel_kernel<<<(BATCH * 32) / 256, 256, 0, stream>>>(vel, Vp);
  prep_w1_kernel<<<64, 256, 0, stream>>>(W1, W1p);
  prep_w2_kernel<<<1024, 256, 0, stream>>>(W2, W2p);
  prep_w3_kernel<<<64, 256, 0, stream>>>(W3, W3p);
  plan_head_kernel<<<BATCH / 64, 256, ldsBytes, stream>>>(Vp, W1p, W2p, W3p,
                                                          b1, b2, b3, gt, acc);
  finalize_kernel<<<1, 1, 0, stream>>>(acc, out);
}
